// EnhancedHyperGeometricMemory_36558761624497
// MI455X (gfx1250) — matrix operand order repaired and checked by reading, not compiled, not run on hardware
//
#include <hip/hip_runtime.h>
#include <hip/hip_bf16.h>

// ---- problem constants ----
#define BB 4
#define SS 4096
#define ROWS (BB*SS)          // 16384
#define INF_ 3.4e38f
#define IND 1024
#define DD 24
#define DP 32                 // padded D
#define TD (3*DD)             // 72
#define MM 16384
#define QQ 8
#define KK 32
#define NSCALES 4
#define SPLIT 2
#define KEYS_PER_SPLIT (MM / SPLIT)   // 8192

typedef _Float16 v16h __attribute__((ext_vector_type(16)));
typedef float    v8f  __attribute__((ext_vector_type(8)));
typedef unsigned int v4u __attribute__((ext_vector_type(4)));
typedef int      v8i  __attribute__((ext_vector_type(8)));
typedef int      v4i  __attribute__((ext_vector_type(4)));

__device__ inline v8f wmma_f16(v16h a, v16h b, v8f c) {
  return __builtin_amdgcn_wmma_f32_16x16x32_f16(false, a, false, b, (short)0, c, false, false);
}

__device__ inline float gelu_exact(float x) {
  return 0.5f * x * (1.0f + erff(x * 0.70710678118654752f));
}

// reduce within each 16-lane segment of the wave32 (xor masks < 16 stay in segment)
__device__ inline float segsum16(float v) {
  v += __shfl_xor(v, 1);
  v += __shfl_xor(v, 2);
  v += __shfl_xor(v, 4);
  v += __shfl_xor(v, 8);
  return v;
}

// ---- Tensor Data Mover: 2-D tile of `rows` rows, `elems8` 8-byte elements per
// row, row stride `stride8` (8-byte units); packed contiguously into LDS.
// D# per CDNA5 ISA groups 0/1 (count=1, type=2, data_size=8B). 6-arg builtin.
__device__ inline void tdm_load_2d(const void* gptr, void* lptr,
                                   unsigned elems8, unsigned rows, unsigned stride8) {
  unsigned long long ga = (unsigned long long)gptr;
  unsigned lds = (unsigned)(unsigned long long)lptr;   // low 32 bits of flat LDS addr = LDS offset
  v4u g0;
  g0.x = 1u;                                           // count=1, user descriptor
  g0.y = lds;                                          // lds_addr
  g0.z = (unsigned)(ga & 0xFFFFFFFFu);                 // global_addr[31:0]
  g0.w = (unsigned)((ga >> 32) & 0x1FFFFFFu) | (2u << 30); // global_addr[56:32] | type=2
  v8i g1;
  g1[0] = (int)(3u << 16);                                    // wg_mask=0, data_size=3 (8B)
  g1[1] = (int)((stride8 & 0xFFFFu) << 16);                   // tensor_dim0[15:0]
  g1[2] = (int)(((stride8 >> 16) & 0xFFFFu) | ((rows & 0xFFFFu) << 16)); // dim0 hi | dim1 lo
  g1[3] = (int)(((rows >> 16) & 0xFFFFu) | ((elems8 & 0xFFFFu) << 16));  // dim1 hi | tile_dim0
  g1[4] = (int)(rows & 0xFFFFu);                              // tile_dim1=rows, tile_dim2=0
  g1[5] = (int)stride8;                                       // tensor_dim0_stride low 32
  g1[6] = 0;
  g1[7] = 0;
  v4i z4 = {0, 0, 0, 0};
  v8i z8 = {0, 0, 0, 0, 0, 0, 0, 0};
  __builtin_amdgcn_tensor_load_to_lds(g0, g1, z4, z4, z8, 0);
}

// =====================================================================
// Kernel 1a: pack keys -> f16 (padded), key norms, phase gain pg[m]
// =====================================================================
__global__ void prep_kernel(const float* __restrict__ keys,
                            const float* __restrict__ qphase,
                            _Float16* __restrict__ keysh,
                            float* __restrict__ kn,
                            float* __restrict__ pg) {
  int m = blockIdx.x * blockDim.x + threadIdx.x;
  if (m >= MM) return;
  float s = 0.f;
  #pragma unroll
  for (int d = 0; d < DD; ++d) {
    float v = keys[m * DD + d];
    keysh[m * DP + d] = (_Float16)v;
    s += v * v;
  }
  #pragma unroll
  for (int d = DD; d < DP; ++d) keysh[m * DP + d] = (_Float16)0.f;
  kn[m] = s;
  float p = 0.f;
  #pragma unroll
  for (int q = 0; q < QQ; ++q) p += tanhf(tanhf(qphase[m * QQ + q]));
  pg[m] = p * (1.0f / QQ);
}

// =====================================================================
// Kernel 1b: weights -> f16, transposed/padded once.
// W_inT_h[n][k] : 80 x 1024 (rows 72..79 zero)   (TDM-friendly layout)
// W_out_h[k][n] : 96 x 1024 (rows 72..95 zero)
// =====================================================================
__global__ void prep_w_kernel(const float* __restrict__ W_in,
                              const float* __restrict__ W_out,
                              _Float16* __restrict__ W_inT_h,
                              _Float16* __restrict__ W_out_h) {
  int i = blockIdx.x * blockDim.x + threadIdx.x;   // grid covers 96*1024
  if (i < 80 * 1024) {
    int n = i >> 10, k = i & 1023;
    W_inT_h[i] = (n < TD) ? (_Float16)W_in[k * TD + n] : (_Float16)0.f;
  }
  if (i < 96 * 1024) {
    int k = i >> 10;
    W_out_h[i] = (k < TD) ? (_Float16)W_out[i] : (_Float16)0.f;
  }
}

// =====================================================================
// Kernel 2: encode  z = GELU(LN(x@W_in+b)) ; q = diag(ricci)*mean3(z)
// block = 128 thr (4 waves), 64 rows; N padded 72->80 (5 col tiles)
// W_in chunks streamed by TDM (double-buffered, overlaps WMMA).
// =====================================================================
__global__ void __launch_bounds__(128, 1)
encode_kernel(const float* __restrict__ x,
              const _Float16* __restrict__ W_inT_h,
              const float* __restrict__ b_in,
              const float* __restrict__ g_in,
              const float* __restrict__ be_in,
              const float* __restrict__ ricci,
              _Float16* __restrict__ qh,
              float* __restrict__ qn) {
  __shared__ _Float16 Wl[2][80][32];   // double-buffered W_in chunk, [n][k]
  __shared__ float    zbuf[64][81];    // post-GELU z tile

  const int tid  = threadIdx.x;
  const int lane = tid & 31;
  const int w    = tid >> 5;           // wave 0..3
  const int m    = lane & 15;
  const int hi   = (lane < 16) ? 0 : 1;
  const int kbA  = hi ? 8 : 0;
  const int kbB  = hi ? 16 : 0;
  const int rowblock = blockIdx.x * 64;
  const int grow = rowblock + w * 16 + m;   // this lane's A row

  v8f acc[5];
  #pragma unroll
  for (int t = 0; t < 5; ++t)
    #pragma unroll
    for (int e = 0; e < 8; ++e) acc[t][e] = 0.f;

  if (tid < 32)
    tdm_load_2d(W_inT_h, &Wl[0][0][0], /*elems8=*/8, /*rows=*/80, /*stride8=*/256);

  for (int c = 0; c < IND / 32; ++c) {
    const int kc = c * 32;
    if (tid < 32) __builtin_amdgcn_s_wait_tensorcnt((short)0);
    __syncthreads();                                  // chunk c visible; buf (c+1)&1 free
    if (tid < 32 && c + 1 < IND / 32)
      tdm_load_2d(W_inT_h + (size_t)(kc + 32), &Wl[(c + 1) & 1][0][0], 8, 80, 256);

    if (kc + 32 < IND) __builtin_prefetch(&x[grow * IND + kc + 32], 0, 1);
    v16h a;
    #pragma unroll
    for (int i = 0; i < 16; ++i) {
      int K = kbB + i;
      a[i] = (_Float16)x[grow * IND + kc + K];
    }
    #pragma unroll
    for (int t = 0; t < 5; ++t) {
      int n = t * 16 + m;
      v16h b;
      #pragma unroll
      for (int i = 0; i < 16; ++i) b[i] = Wl[c & 1][n][kbB + i];
      acc[t] = wmma_f16(a, b, acc[t]);
    }
  }

  // bias + LayerNorm(72) + GELU, write z tile to LDS
  #pragma unroll
  for (int v = 0; v < 8; ++v) {
    float s = 0.f, s2 = 0.f, vals[5];
    #pragma unroll
    for (int t = 0; t < 5; ++t) {
      int col = t * 16 + m;
      bool valid = col < TD;
      float z = acc[t][v] + (valid ? b_in[col] : 0.f);
      vals[t] = z;
      if (valid) { s += z; s2 += z * z; }
    }
    s  = segsum16(s);
    s2 = segsum16(s2);
    float mean = s * (1.f / TD);
    float var  = s2 * (1.f / TD) - mean * mean;
    float rs   = rsqrtf(var + 1e-5f);
    int lrow = w * 16 + v + hi * 8;
    #pragma unroll
    for (int t = 0; t < 5; ++t) {
      int col = t * 16 + m;
      if (col < TD) {
        float zn = (vals[t] - mean) * rs * g_in[col] + be_in[col];
        zbuf[lrow][col] = gelu_exact(zn);
      }
    }
  }
  __syncthreads();

  // q = diag(ricci) * mean over 3 components; f16-pack + norm
  int r = tid >> 1, h = tid & 1;
  float part = 0.f;
  #pragma unroll
  for (int j = 0; j < 12; ++j) {
    int d = h * 12 + j;
    float qv = (zbuf[r][3 * d] + zbuf[r][3 * d + 1] + zbuf[r][3 * d + 2]) *
               (1.f / 3.f) * ricci[d * (DD + 1)];
    qh[(rowblock + r) * DP + d] = (_Float16)qv;
    part += qv * qv;
  }
  part += __shfl_xor(part, 1);
  if (h == 0) {
    qn[rowblock + r] = part;
  } else {
    #pragma unroll
    for (int p = DD; p < DP; ++p) qh[(rowblock + r) * DP + p] = (_Float16)0.f;
  }
}

// =====================================================================
// Kernel 3: split-K distance + partial top-32 (sorted, raw distances).
// Grid = (ROWS/128) * SPLIT blocks; block = 128 thr = 128 queries,
// scanning KEYS_PER_SPLIT keys.  Key chunks streamed by TDM,
// double-buffered; each B fragment loaded once and reused by both
// row-tile WMMAs (ct outer, rt2 inner keeps live VGPRs low).
// =====================================================================
__global__ void __launch_bounds__(128, 1)
dist_part_kernel(const _Float16* __restrict__ qh,
                 const float* __restrict__ qn,
                 const _Float16* __restrict__ keysh,
                 const float* __restrict__ kn,
                 float* __restrict__ pdist,
                 int* __restrict__ pidx) {
  __shared__ _Float16 kbuf[2][64][DP]; // double-buffered key chunk (f16, padded)
  __shared__ float    dbuf[128][65];   // distance tile (+1 pad vs banks)
  __shared__ float    knbuf[2][64];
  __shared__ float    qnbuf[128];

  const int tid  = threadIdx.x;
  const int lane = tid & 31;
  const int w    = tid >> 5;
  const int m    = lane & 15;
  const int hi   = (lane < 16) ? 0 : 1;
  const int kbA  = hi ? 8 : 0;
  const int kbB  = hi ? 16 : 0;
  const int qblk  = blockIdx.x / SPLIT;
  const int split = blockIdx.x % SPLIT;
  const int qbase = qblk * 128;
  const int kbase = split * KEYS_PER_SPLIT;

  qnbuf[tid] = qn[qbase + tid];

  float td[KK];
  int   ti[KK];
  #pragma unroll
  for (int j = 0; j < KK; ++j) { td[j] = INF_; ti[j] = 0; }

  // A fragments: loaded once, reused for all key chunks
  v16h afrag[2];
  #pragma unroll
  for (int rt2 = 0; rt2 < 2; ++rt2) {
    int row = qbase + (w * 2 + rt2) * 16 + m;
    #pragma unroll
    for (int i = 0; i < 16; ++i) {
      int K = kbB + i;
      afrag[rt2][i] = qh[row * DP + K];
    }
  }

  if (tid < 32)
    tdm_load_2d(keysh + (size_t)kbase * DP, &kbuf[0][0][0], 512, 1, 512);
  if (tid < 64) knbuf[0][tid] = kn[kbase + tid];

  const int NCHUNK = KEYS_PER_SPLIT / 64;
  for (int chunk = 0; chunk < NCHUNK; ++chunk) {
    const int k0 = kbase + chunk * 64;
    const int cur = chunk & 1;
    if (tid < 32) __builtin_amdgcn_s_wait_tensorcnt((short)0);
    __syncthreads();                                  // chunk data visible; prev dbuf consumed
    if (chunk + 1 < NCHUNK) {
      if (tid < 32)
        tdm_load_2d(keysh + (size_t)(k0 + 64) * DP, &kbuf[cur ^ 1][0][0], 512, 1, 512);
      if (tid < 64) knbuf[cur ^ 1][tid] = kn[k0 + 64 + tid];
    }

    // ct outer: load each B fragment once, reuse for both row tiles
    #pragma unroll
    for (int ct = 0; ct < 4; ++ct) {
      int n = ct * 16 + m;
      v16h b;
      #pragma unroll
      for (int i = 0; i < 16; ++i) b[i] = kbuf[cur][n][kbB + i];
      float knv = knbuf[cur][ct * 16 + m];
      #pragma unroll
      for (int rt2 = 0; rt2 < 2; ++rt2) {
        int rt = w * 2 + rt2;
        v8f c = {0.f, 0.f, 0.f, 0.f, 0.f, 0.f, 0.f, 0.f};
        c = wmma_f16(afrag[rt2], b, c);
        #pragma unroll
        for (int v = 0; v < 8; ++v) {
          int lrow = rt * 16 + v + hi * 8;
          float d2 = qnbuf[lrow] + knv - 2.f * c[v];
          dbuf[lrow][ct * 16 + m] = sqrtf(fmaxf(d2, 0.f));
        }
      }
    }
    __syncthreads();

    // per-query top-K: sorted insertion, all-static indices -> VGPRs
    #pragma unroll 1
    for (int j = 0; j < 64; ++j) {
      float dv = dbuf[tid][j];
      if (dv < td[KK - 1]) {
        td[KK - 1] = dv;
        ti[KK - 1] = k0 + j;
        #pragma unroll
        for (int s = KK - 1; s > 0; --s) {
          if (td[s] < td[s - 1]) {
            float tf = td[s]; td[s] = td[s - 1]; td[s - 1] = tf;
            int   tx = ti[s]; ti[s] = ti[s - 1]; ti[s - 1] = tx;
          }
        }
      }
    }
  }

  // emit sorted partial lists (raw distances)
  const int row = qbase + tid;
  float* pd = pdist + ((size_t)row * SPLIT + split) * KK;
  int*   pi = pidx  + ((size_t)row * SPLIT + split) * KK;
  #pragma unroll
  for (int j = 0; j < KK; ++j) { pd[j] = td[j]; pi[j] = ti[j]; }
}

// =====================================================================
// Kernel 3b: merge SPLIT sorted partial lists -> final top-32, softmax.
// =====================================================================
__global__ void __launch_bounds__(256, 1)
merge_softmax_kernel(const float* __restrict__ pdist,
                     const int* __restrict__ pidx,
                     const float* __restrict__ pg,
                     const float* __restrict__ fractal_w,
                     const float* __restrict__ temperature,
                     float* __restrict__ wgt,
                     int* __restrict__ idxb) {
  const int row = blockIdx.x * blockDim.x + threadIdx.x;
  if (row >= ROWS) return;
  const float* pd = pdist + (size_t)row * (SPLIT * KK);
  const int*   pi = pidx  + (size_t)row * (SPLIT * KK);

  float td[KK];
  int   ti[KK];
  #pragma unroll
  for (int j = 0; j < KK; ++j) { td[j] = pd[j]; ti[j] = pi[j]; }

  #pragma unroll 1
  for (int j = KK; j < SPLIT * KK; ++j) {   // stream remaining sorted candidates
    float dv = pd[j];
    if (dv < td[KK - 1]) {
      td[KK - 1] = dv;
      ti[KK - 1] = pi[j];
      #pragma unroll
      for (int s = KK - 1; s > 0; --s) {
        if (td[s] < td[s - 1]) {
          float tf = td[s]; td[s] = td[s - 1]; td[s - 1] = tf;
          int   tx = ti[s]; ti[s] = ti[s - 1]; ti[s - 1] = tx;
        }
      }
    }
  }

  // fractal scale = sum softmax(fractal_w)[s] * 2^-s ; temperature
  float fw0 = fractal_w[0], fw1 = fractal_w[1], fw2 = fractal_w[2], fw3 = fractal_w[3];
  float fm = fmaxf(fmaxf(fw0, fw1), fmaxf(fw2, fw3));
  float e0 = expf(fw0 - fm), e1 = expf(fw1 - fm), e2 = expf(fw2 - fm), e3 = expf(fw3 - fm);
  float scale = (e0 + 0.5f * e1 + 0.25f * e2 + 0.125f * e3) / (e0 + e1 + e2 + e3);
  float invT = scale / fmaxf(temperature[0], 1e-6f);

  float lm = -INF_;
  #pragma unroll
  for (int j = 0; j < KK; ++j) {
    float lg = -(td[j] * invT) + 0.05f * pg[ti[j]];
    td[j] = lg;
    lm = fmaxf(lm, lg);
  }
  float Z = 0.f;
  #pragma unroll
  for (int j = 0; j < KK; ++j) { td[j] = expf(td[j] - lm); Z += td[j]; }
  float rZ = 1.f / Z;
  #pragma unroll
  for (int j = 0; j < KK; ++j) {
    wgt[row * KK + j]  = td[j] * rZ;
    idxb[row * KK + j] = ti[j];
  }
}

// =====================================================================
// Kernel 4: read = sum_k w_k * values[idx_k]   (one wave per row)
// =====================================================================
__global__ void gather_kernel(const float* __restrict__ values,
                              const float* __restrict__ wgt,
                              const int* __restrict__ idxb,
                              float* __restrict__ readb) {
  const int lane = threadIdx.x & 31;
  const int w    = threadIdx.x >> 5;
  const int row  = blockIdx.x * 8 + w;
  float a0 = 0.f, a1 = 0.f, a2 = 0.f;
  #pragma unroll 4
  for (int k = 0; k < KK; ++k) {
    float wk = wgt[row * KK + k];
    const float* vp = values + (size_t)idxb[row * KK + k] * TD;
    a0 += wk * vp[lane];
    a1 += wk * vp[lane + 32];
    if (lane < 8) a2 += wk * vp[lane + 64];
  }
  readb[row * TD + lane]      = a0;
  readb[row * TD + lane + 32] = a1;
  if (lane < 8) readb[row * TD + lane + 64] = a2;
}

// =====================================================================
// Kernel 5: out = GELU(LN(read@W_out + b_out))
// block = 256 thr (8 waves), 16 rows x 1024 cols; LN via LDS ds_add_f32.
// A from readb (guard folds per unrolled chunk), B from padded f16 W_out_h.
// =====================================================================
__global__ void __launch_bounds__(256, 1)
output_kernel(const float* __restrict__ readb,
              const _Float16* __restrict__ W_out_h,
              const float* __restrict__ b_out,
              const float* __restrict__ g_out,
              const float* __restrict__ be_out,
              float* __restrict__ out) {
  __shared__ float rowSum[16], rowSq[16];

  const int tid  = threadIdx.x;
  const int lane = tid & 31;
  const int w    = tid >> 5;          // wave 0..7
  const int m    = lane & 15;
  const int hi   = (lane < 16) ? 0 : 1;
  const int kbA  = hi ? 8 : 0;
  const int kbB  = hi ? 16 : 0;
  const int r0   = blockIdx.x * 16;

  if (tid < 16) { rowSum[tid] = 0.f; rowSq[tid] = 0.f; }
  __syncthreads();

  v8f acc[8];
  #pragma unroll
  for (int t = 0; t < 8; ++t)
    #pragma unroll
    for (int e = 0; e < 8; ++e) acc[t][e] = 0.f;

  #pragma unroll
  for (int kc = 0; kc < 96; kc += 32) {
    v16h a;
    #pragma unroll
    for (int i = 0; i < 16; ++i) {
      int K = kc + kbB + i;
      a[i] = (K < TD) ? (_Float16)readb[(r0 + m) * TD + K] : (_Float16)0.f;
    }
    #pragma unroll
    for (int t = 0; t < 8; ++t) {
      int col = (w * 8 + t) * 16 + m;
      v16h b;
      #pragma unroll
      for (int i = 0; i < 16; ++i) b[i] = W_out_h[(kc + kbB + i) * IND + col];
      acc[t] = wmma_f16(a, b, acc[t]);
    }
  }

  // bias + cross-wave LN reduction via LDS float atomics
  #pragma unroll
  for (int v = 0; v < 8; ++v) {
    float s = 0.f, s2 = 0.f;
    #pragma unroll
    for (int t = 0; t < 8; ++t) {
      int col = (w * 8 + t) * 16 + m;
      float z = acc[t][v] + b_out[col];
      acc[t][v] = z;
      s  += z;
      s2 += z * z;
    }
    s  = segsum16(s);
    s2 = segsum16(s2);
    if (m == 0) {                      // lane 0 (rows v) and lane 16 (rows v+8)
      atomicAdd(&rowSum[v + hi * 8], s);
      atomicAdd(&rowSq[v + hi * 8], s2);
    }
  }
  __syncthreads();

  #pragma unroll
  for (int v = 0; v < 8; ++v) {
    int vrow = v + hi * 8;
    float mean = rowSum[vrow] * (1.f / IND);
    float var  = rowSq[vrow] * (1.f / IND) - mean * mean;
    float rs   = rsqrtf(var + 1e-5f);
    int row = r0 + vrow;
    #pragma unroll
    for (int t = 0; t < 8; ++t) {
      int col = (w * 8 + t) * 16 + m;
      float zn = (acc[t][v] - mean) * rs * g_out[col] + be_out[col];
      out[(size_t)row * IND + col] = gelu_exact(zn);
    }
  }
}

// =====================================================================
extern "C" void kernel_launch(void* const* d_in, const int* in_sizes, int n_in,
                              void* d_out, int out_size, void* d_ws, size_t ws_size,
                              hipStream_t stream) {
  (void)in_sizes; (void)n_in; (void)out_size; (void)ws_size;
  const float* x        = (const float*)d_in[0];
  const float* W_in     = (const float*)d_in[1];
  const float* b_in     = (const float*)d_in[2];
  const float* g_in     = (const float*)d_in[3];
  const float* be_in    = (const float*)d_in[4];
  const float* ricci    = (const float*)d_in[5];
  const float* keys     = (const float*)d_in[6];
  const float* values   = (const float*)d_in[7];
  const float* qphase   = (const float*)d_in[8];
  const float* fractalw = (const float*)d_in[9];
  const float* W_out    = (const float*)d_in[10];
  const float* b_out    = (const float*)d_in[11];
  const float* g_out    = (const float*)d_in[12];
  const float* be_out   = (const float*)d_in[13];
  const float* temp     = (const float*)d_in[14];

  char* ws = (char*)d_ws;
  const size_t MB = 1024ull * 1024ull;
  _Float16* qh      = (_Float16*)(ws + 0);                 // 1 MiB
  _Float16* keysh   = (_Float16*)(ws + 1 * MB);            // 1 MiB
  float*    qn      = (float*)(ws + 2 * MB);               // 64 KiB
  float*    kn      = (float*)(ws + 2 * MB + 64 * 1024);   // 64 KiB
  float*    pg      = (float*)(ws + 2 * MB + 128 * 1024);  // 64 KiB
  float*    wgt     = (float*)(ws + 2 * MB + 192 * 1024);  // 2 MiB
  int*      idxb    = (int*)(ws + 4 * MB + 192 * 1024);    // 2 MiB
  float*    readb   = (float*)(ws + 6 * MB + 192 * 1024);  // 4.5 MiB
  _Float16* W_inT_h = (_Float16*)(ws + 11 * MB);           // 160 KiB
  _Float16* W_out_h = (_Float16*)(ws + 11 * MB + 256 * 1024); // 192 KiB
  float*    pdist   = (float*)(ws + 12 * MB);              // 4 MiB
  int*      pidx    = (int*)(ws + 16 * MB);                // 4 MiB

  prep_kernel<<<MM / 256, 256, 0, stream>>>(keys, qphase, keysh, kn, pg);
  prep_w_kernel<<<(96 * 1024) / 256, 256, 0, stream>>>(W_in, W_out, W_inT_h, W_out_h);
  encode_kernel<<<ROWS / 64, 128, 0, stream>>>(x, W_inT_h, b_in, g_in, be_in, ricci, qh, qn);
  dist_part_kernel<<<(ROWS / 128) * SPLIT, 128, 0, stream>>>(qh, qn, keysh, kn, pdist, pidx);
  merge_softmax_kernel<<<ROWS / 256, 256, 0, stream>>>(pdist, pidx, pg, fractalw, temp,
                                                       wgt, idxb);
  gather_kernel<<<ROWS / 8, 256, 0, stream>>>(values, wgt, idxb, readb);
  output_kernel<<<ROWS / 16, 256, 0, stream>>>(readb, W_out_h, b_out, g_out, be_out,
                                               (float*)d_out);
}
